// DotProductAttention_27101243638408
// MI455X (gfx1250) — 8-bit matrix operand order repaired and checked by reading, compiled, not run on hardware
//
#include <hip/hip_runtime.h>

#define B_    16
#define S_    2048
#define D_    128
#define QTILE 128
#define KTILE 64
#define NWAVE 8

typedef __attribute__((ext_vector_type(16))) int          v16i;
typedef __attribute__((ext_vector_type(8)))  int          v8i;
typedef __attribute__((ext_vector_type(8)))  float        v8f;
typedef __attribute__((ext_vector_type(4)))  float        f32x4;
typedef __attribute__((ext_vector_type(4)))  unsigned int u32x4;
typedef __attribute__((ext_vector_type(2)))  unsigned int u32x2;

union Frag16 {
    v16i i;
    unsigned int d[16];
    u32x4 q[4];
};
union Frag8 {
    v8i i;
    unsigned int d[8];
    u32x2 p[4];
    u32x4 q[2];
};

static __device__ __forceinline__ unsigned int f2fp8_sw(float x) {
    unsigned int u = __float_as_uint(x);
    unsigned int sgn = (u >> 24) & 0x80u;
    float ax = __builtin_fabsf(x);
    if (!(ax > 0.0f)) return sgn;
    ax = fminf(ax, 448.0f);
    unsigned int b = __float_as_uint(ax);
    int e = (int)((b >> 23) & 0xFFu) - 127;
    if (e < -6) {
        unsigned int m = (unsigned int)(ax * 512.0f + 0.5f);
        if (m > 7u) m = 7u;
        return sgn | m;
    }
    unsigned int mant = (b >> 20) & 7u;
    unsigned int rem  = b & 0xFFFFFu;
    if (rem > 0x80000u || (rem == 0x80000u && (mant & 1u))) {
        mant++;
        if (mant == 8u) { mant = 0u; ++e; }
    }
    if (e > 8) { e = 8; mant = 6u; }
    return sgn | ((unsigned int)(e + 7) << 3) | mant;
}

#if defined(__has_builtin)
#if __has_builtin(__builtin_amdgcn_cvt_pk_fp8_f32)
#define HAVE_HW_FP8_CVT 1
#endif
#endif

static __device__ __forceinline__ unsigned int pack4_fp8(float f0, float f1,
                                                         float f2, float f3) {
#ifdef HAVE_HW_FP8_CVT
    unsigned int w = (unsigned int)__builtin_amdgcn_cvt_pk_fp8_f32(f0, f1, 0, false);
    w = (unsigned int)__builtin_amdgcn_cvt_pk_fp8_f32(f2, f3, (int)w, true);
    return w;
#else
    return f2fp8_sw(f0) | (f2fp8_sw(f1) << 8) |
           (f2fp8_sw(f2) << 16) | (f2fp8_sw(f3) << 24);
#endif
}

static __device__ __forceinline__ unsigned char fp8_byte(float f) {
#ifdef HAVE_HW_FP8_CVT
    return (unsigned char)(((unsigned int)
        __builtin_amdgcn_cvt_pk_fp8_f32(f, 0.0f, 0, false)) & 0xFFu);
#else
    return (unsigned char)f2fp8_sw(f);
#endif
}

static __device__ __forceinline__ float half_reduce_max(float v) {
#pragma unroll
    for (int m = 1; m < 16; m <<= 1) v = fmaxf(v, __shfl_xor(v, m, 16));
    return v;
}
static __device__ __forceinline__ float half_reduce_sum(float v) {
#pragma unroll
    for (int m = 1; m < 16; m <<= 1) v += __shfl_xor(v, m, 16);
    return v;
}

__global__ __launch_bounds__(256, 1)
void gated_attn_fwd_fp8(const float* __restrict__ Q, const float* __restrict__ K,
                        const float* __restrict__ V, const float* __restrict__ PQ,
                        const float* __restrict__ PK, float* __restrict__ Out) {
    __shared__ __align__(16) unsigned char sK[KTILE * D_];
    __shared__ __align__(16) unsigned char sVt[D_ * KTILE];
    __shared__ __align__(16) unsigned char sP[NWAVE][16 * KTILE];
    __shared__ float sPk[KTILE];

    const int tid  = threadIdx.x;
    const int wave = tid >> 5;
    const int lane = tid & 31;
    const int l16  = lane & 15;
    const int half = lane >> 4;

    const int blocksPerB = S_ / QTILE;
    const int b     = blockIdx.x / blocksPerB;
    const int qbase = (blockIdx.x % blocksPerB) * QTILE;

    const float* Qb  = Q  + (size_t)b * S_ * D_;
    const float* Kb  = K  + (size_t)b * S_ * D_;
    const float* Vb  = V  + (size_t)b * S_ * D_;
    const float* pqb = PQ + (size_t)b * S_;
    const float* pkb = PK + (size_t)b * S_;
    float*       Ob  = Out + (size_t)b * S_ * D_;

    const float SCALE = 0.088388347648318447f;

    const int qrow = qbase + wave * 16 + l16;
    Frag16 aQ;
#pragma unroll
    for (int g = 0; g < 8; ++g) {
        const float* qp = Qb + qrow * D_ + (g >> 1) * 32 + half * 16 + (g & 1) * 8;
        f32x4 x0 = *(const f32x4*)(qp);
        f32x4 x1 = *(const f32x4*)(qp + 4);
        aQ.d[2 * g]     = pack4_fp8(x0[0], x0[1], x0[2], x0[3]);
        aQ.d[2 * g + 1] = pack4_fp8(x1[0], x1[1], x1[2], x1[3]);
    }

    float pqv[8];
#pragma unroll
    for (int i = 0; i < 8; ++i)
        pqv[i] = pqb[qbase + wave * 16 + half * 8 + i];

    float rmax[8], rsum[8];
    v8f Oacc[8];
#pragma unroll
    for (int i = 0; i < 8; ++i) { rmax[i] = -__builtin_inff(); rsum[i] = 0.0f; }
#pragma unroll
    for (int d = 0; d < 8; ++d)
#pragma unroll
        for (int j = 0; j < 8; ++j) Oacc[d][j] = 0.0f;

    const int nkt = S_ / KTILE;
    for (int kt = 0; kt < nkt; ++kt) {
        __syncthreads();
#pragma unroll
        for (int it = 0; it < 4; ++it) {
            int idx = tid + it * 256;
            int key = idx >> 4;
            int d0  = (idx & 15) << 3;
            const float* kp = Kb + (kt * KTILE + key) * D_ + d0;
            f32x4 k0 = *(const f32x4*)(kp);
            f32x4 k1 = *(const f32x4*)(kp + 4);
            u32x2 kw;
            kw.x = pack4_fp8(k0[0], k0[1], k0[2], k0[3]);
            kw.y = pack4_fp8(k1[0], k1[1], k1[2], k1[3]);
            *(u32x2*)&sK[key * D_ + d0] = kw;

            const float* vp = Vb + (kt * KTILE + key) * D_ + d0;
            f32x4 v0 = *(const f32x4*)(vp);
            f32x4 v1 = *(const f32x4*)(vp + 4);
#pragma unroll
            for (int j = 0; j < 4; ++j) {
                sVt[(d0 + j) * KTILE + key]     = fp8_byte(v0[j]);
                sVt[(d0 + 4 + j) * KTILE + key] = fp8_byte(v1[j]);
            }
        }
        if (tid < KTILE) sPk[tid] = pkb[kt * KTILE + tid];
        if (kt + 1 < nkt) {
            __builtin_prefetch((const char*)(Kb + (kt + 1) * KTILE * D_) + tid * 128, 0, 0);
            __builtin_prefetch((const char*)(Vb + (kt + 1) * KTILE * D_) + tid * 128, 0, 0);
        }
        __syncthreads();

        float sc[4][8];
        float tmax[8];
#pragma unroll
        for (int i = 0; i < 8; ++i) tmax[i] = -__builtin_inff();
#pragma unroll
        for (int nb = 0; nb < 4; ++nb) {
            Frag16 bK;
#pragma unroll
            for (int g = 0; g < 4; ++g)
                bK.q[g] = *(const u32x4*)&sK[(nb * 16 + l16) * D_ + g * 32 + half * 16];
            v8f acc;
#pragma unroll
            for (int j = 0; j < 8; ++j) acc[j] = 0.0f;
            acc = __builtin_amdgcn_wmma_f32_16x16x128_fp8_fp8(
                aQ.i, bK.i, (short)0, acc, false, false);
            float pkv = sPk[nb * 16 + l16];
#pragma unroll
            for (int i = 0; i < 8; ++i) {
                float s = acc[i] * SCALE * pqv[i] * pkv;
                sc[nb][i] = s;
                tmax[i] = fmaxf(tmax[i], s);
            }
        }

        float fac[8];
#pragma unroll
        for (int i = 0; i < 8; ++i) {
            float tm = half_reduce_max(tmax[i]);
            float nm = fmaxf(rmax[i], tm);
            fac[i] = __expf(rmax[i] - nm);
            rmax[i] = nm;
        }
        float rs[8];
#pragma unroll
        for (int i = 0; i < 8; ++i) rs[i] = 0.0f;
#pragma unroll
        for (int nb = 0; nb < 4; ++nb)
#pragma unroll
            for (int i = 0; i < 8; ++i) {
                float e = __expf(sc[nb][i] - rmax[i]);
                rs[i] += e;
                sP[wave][(half * 8 + i) * KTILE + nb * 16 + l16] = fp8_byte(e);
            }
#pragma unroll
        for (int i = 0; i < 8; ++i)
            rsum[i] = rsum[i] * fac[i] + half_reduce_sum(rs[i]);
#pragma unroll
        for (int d = 0; d < 8; ++d)
#pragma unroll
            for (int i = 0; i < 8; ++i) Oacc[d][i] *= fac[i];

        asm volatile("s_wait_dscnt 0" ::: "memory");

        Frag8 aP;
#pragma unroll
        for (int g = 0; g < 4; ++g)
            aP.p[g] = *(const u32x2*)&sP[wave][l16 * KTILE + (g >> 1) * 32 + half * 16 + (g & 1) * 8];

#pragma unroll
        for (int d = 0; d < 8; ++d) {
            Frag8 bV;
            bV.q[0] = *(const u32x4*)&sVt[(d * 16 + l16) * KTILE + half * 16];
            bV.q[1] = *(const u32x4*)&sVt[(d * 16 + l16) * KTILE + 32 + half * 16];
            Oacc[d] = __builtin_amdgcn_wmma_f32_16x16x64_fp8_fp8(
                aP.i, bV.i, (short)0, Oacc[d], false, false);
        }
    }

    float inv[8];
#pragma unroll
    for (int i = 0; i < 8; ++i) inv[i] = 1.0f / rsum[i];
#pragma unroll
    for (int d = 0; d < 8; ++d)
#pragma unroll
        for (int i = 0; i < 8; ++i) {
            int row = qbase + wave * 16 + half * 8 + i;
            Ob[row * D_ + d * 16 + l16] = Oacc[d][i] * inv[i];
        }
}

extern "C" void kernel_launch(void* const* d_in, const int* in_sizes, int n_in,
                              void* d_out, int out_size, void* d_ws, size_t ws_size,
                              hipStream_t stream) {
    const float* Q  = (const float*)d_in[0];
    const float* K  = (const float*)d_in[1];
    const float* V  = (const float*)d_in[2];
    const float* PQ = (const float*)d_in[3];
    const float* PK = (const float*)d_in[4];
    float* Out = (float*)d_out;

    dim3 grid(B_ * (S_ / QTILE));
    dim3 block(256);
    gated_attn_fwd_fp8<<<grid, block, 0, stream>>>(Q, K, V, PQ, PK, Out);
}
